// GraphConstructor_36756330119760
// MI455X (gfx1250) — matrix operand order repaired and checked by reading, not compiled, not run on hardware
//
#include <hip/hip_runtime.h>
#include <hip/hip_bf16.h>

// ---------------------------------------------------------------------------
// GraphConstructor for MI455X (gfx1250, wave32, WMMA)
//
//   H (4,4096,512) f32  ->  A_norm (4,4096,4096) f32
//
// Math collapse: row degrees of A are provably always 11 (10 distinct top-k
// hits set to 1, plus eye), so A_norm = A / (11+1e-8). Only the top-k indices
// matter; everything else is a constant scatter over a zeroed dense output.
//
// Kernel 1: row L2-normalize H, emit bf16 Hn into workspace (16 MiB, L2-resident).
// Kernel 2: per 16-row strip: zero output rows, bf16 WMMA GEMM strip vs all
//           4096 columns (chunked through LDS), streaming per-thread top-10,
//           merge, scatter c2 / diagonal add.
// ---------------------------------------------------------------------------

typedef __attribute__((ext_vector_type(16))) __bf16 v16bf;
typedef __attribute__((ext_vector_type(8)))  float  v8f;

#define N_PTS 4096
#define DIM   512
#define KNB   10
#define NC    512          // sim columns per LDS chunk
#define LDSTR 17           // padded floats per sim column (bank-conflict-free scan)

union Frag { v16bf v; uint4 q[2]; };

__device__ __forceinline__ unsigned bf16rne(float f) {
    unsigned x = __float_as_uint(f);
    unsigned r = x + 0x7FFFu + ((x >> 16) & 1u);   // round-to-nearest-even
    return r >> 16;
}

// ---------------- Kernel 1: normalize rows, f32 -> bf16 ----------------
__global__ __launch_bounds__(256) void gc_normalize(const float* __restrict__ H,
                                                    unsigned short* __restrict__ hn) {
    const int tid = threadIdx.x;
    const int ln  = tid & 31;
    const int wv  = tid >> 5;
    const int row = blockIdx.x * 8 + wv;           // 16384 rows total
    const float4* h4 = (const float4*)(H + (size_t)row * DIM);

    float4 d[4];
    float  s = 0.f;
#pragma unroll
    for (int j = 0; j < 4; ++j) {
        d[j] = h4[ln + 32 * j];
        s += d[j].x * d[j].x + d[j].y * d[j].y + d[j].z * d[j].z + d[j].w * d[j].w;
    }
#pragma unroll
    for (int off = 16; off > 0; off >>= 1)
        s += __shfl_xor(s, off, 32);

    const float inv = 1.0f / fmaxf(sqrtf(s), 1e-12f);

    unsigned short* orow = hn + (size_t)row * DIM;
#pragma unroll
    for (int j = 0; j < 4; ++j) {
        uint2 p;
        p.x = bf16rne(d[j].x * inv) | (bf16rne(d[j].y * inv) << 16);
        p.y = bf16rne(d[j].z * inv) | (bf16rne(d[j].w * inv) << 16);
        *(uint2*)(orow + (ln + 32 * j) * 4) = p;
    }
}

// ------ Kernel 2: WMMA GEMM strip + streaming top-k + scatter output ------
__global__ __launch_bounds__(256) void gc_strip(const unsigned short* __restrict__ hn,
                                                float* __restrict__ out) {
    __shared__ float lds[NC * LDSTR];              // 34,816 B sim chunk
    float* candV = lds;                            // merge phase aliases (16*16*10 f32)
    int*   candI = (int*)(lds + 16 * 16 * KNB);    // + (16*16*10 i32), fits in 34 KB

    const int tid = threadIdx.x;
    const int ln  = tid & 31;                      // lane in wave32
    const int wv  = tid >> 5;                      // wave 0..7
    const int b   = blockIdx.x >> 8;               // batch 0..3
    const int mt  = blockIdx.x & 255;              // 16-row m-tile 0..255

    const unsigned short* hb = hn + (size_t)b * N_PTS * DIM;
    float* outRows = out + ((size_t)b * N_PTS + (size_t)mt * 16) * N_PTS;

    // ---- zero-fill the 16 owned output rows (256 KiB) ----
    {
        float4  z  = make_float4(0.f, 0.f, 0.f, 0.f);
        float4* o4 = (float4*)outRows;
#pragma unroll
        for (int i = 0; i < (16 * N_PTS / 4) / 256; ++i)
            o4[tid + 256 * i] = z;
    }
    __threadfence();   // zeros visible before the final scatter read-modify-write

    // ---- per-thread top-10 (registers, fully unrolled indexing) ----
    float tv[KNB]; int ti[KNB];
#pragma unroll
    for (int j = 0; j < KNB; ++j) { tv[j] = -1e30f; ti[j] = -1; }

    // A-fragment addressing (16x32 bf16): lanes 0-15 carry K {0..7,16..23},
    // lanes 16-31 carry K {8..15,24..31}; two 16B loads per lane per k-step.
    const int rowA = mt * 16 + (ln & 15);
    const unsigned short* aRow = hb + (size_t)rowA * DIM + ((ln & 16) ? 8 : 0);

    const int r = tid >> 4;                        // top-k row 0..15
    const int s = tid & 15;                        // sub-column 0..15

    for (int chunk = 0; chunk < N_PTS / NC; ++chunk) {   // 8 chunks of 512 cols
        const int cbase = chunk * NC;

        // -- WMMA: each wave computes 4 n-tiles of this chunk --
#pragma unroll 1
        for (int u = 0; u < NC / 16 / 8; ++u) {
            const int nloc = wv * (NC / 16 / 8) + u;        // 0..31
            // B-fragment (32x16 bf16): column n = row n of Hn; lanes 0-15
            // carry K {0..7,16..23}, lanes 16-31 carry K {8..15,24..31}.
            const int rowB = cbase + nloc * 16 + (ln & 15);
            const unsigned short* bRow = hb + (size_t)rowB * DIM + ((ln & 16) ? 8 : 0);

            v8f acc = {0.f, 0.f, 0.f, 0.f, 0.f, 0.f, 0.f, 0.f};
#pragma unroll
            for (int kk = 0; kk < DIM / 32; ++kk) {         // 16 k-steps
                Frag fa, fb;
                const unsigned short* pa = aRow + kk * 32;
                fa.q[0] = *(const uint4*)(pa);
                fa.q[1] = *(const uint4*)(pa + 16);
                const unsigned short* pb = bRow + kk * 32;
                fb.q[0] = *(const uint4*)(pb);
                fb.q[1] = *(const uint4*)(pb + 16);
                acc = __builtin_amdgcn_wmma_f32_16x16x32_bf16(
                        false, fa.v, false, fb.v, (short)0, acc, false, false);
            }
            // C layout: VGPR j, lanes 0-15 -> (M=j, N=ln), lanes 16-31 -> (M=j+8, N=ln-16).
            const int colB  = nloc * 16 + (ln & 15);
            const int rbase = (ln & 16) ? 8 : 0;
#pragma unroll
            for (int j = 0; j < 8; ++j)
                lds[colB * LDSTR + rbase + j] = acc[j];
        }
        __syncthreads();

        // -- streaming top-10 scan: 16 threads per row, strided columns --
#pragma unroll 1
        for (int i = 0; i < NC / 16; ++i) {
            const int   cl = s + 16 * i;
            const float v  = lds[cl * LDSTR + r];
            if (v > tv[KNB - 1]) {
                tv[KNB - 1] = v; ti[KNB - 1] = cbase + cl;
#pragma unroll
                for (int j = KNB - 1; j > 0; --j) {
                    if (tv[j] > tv[j - 1]) {
                        float tf = tv[j]; tv[j] = tv[j - 1]; tv[j - 1] = tf;
                        int   tt = ti[j]; ti[j] = ti[j - 1]; ti[j - 1] = tt;
                    }
                }
            }
        }
        __syncthreads();
    }

    // ---- merge the 16 per-row candidate lists ----
#pragma unroll
    for (int j = 0; j < KNB; ++j) {
        candV[(r * 16 + s) * KNB + j] = tv[j];
        candI[(r * 16 + s) * KNB + j] = ti[j];
    }
    __syncthreads();

    if (tid < 16) {
        const int   rr   = tid;
        float*      cV   = candV + rr * 16 * KNB;       // 160 candidates
        const int*  cI   = candI + rr * 16 * KNB;
        const int   grow = mt * 16 + rr;
        float*      orow = outRows + (size_t)rr * N_PTS;
        const float c2   = 1.0f / (11.0f + 1e-8f);      // deg == 11 for every row

#pragma unroll 1
        for (int k = 0; k < KNB; ++k) {
            float best = -1e38f; int bi = 0;
            for (int q = 0; q < 16 * KNB; ++q) {
                float v = cV[q];
                if (v > best) { best = v; bi = q; }
            }
            cV[bi] = -1e38f;
            orow[cI[bi]] = c2;                           // top-k scatter
        }
        orow[grow] = orow[grow] + c2;                    // + eye (same-wave ordered)
    }
}

extern "C" void kernel_launch(void* const* d_in, const int* in_sizes, int n_in,
                              void* d_out, int out_size, void* d_ws, size_t ws_size,
                              hipStream_t stream) {
    const float* H   = (const float*)d_in[0];
    float*       out = (float*)d_out;
    unsigned short* hn = (unsigned short*)d_ws;          // 16 MiB bf16 Hn

    gc_normalize<<<(4 * N_PTS) / 8, 256, 0, stream>>>(H, hn);
    gc_strip<<<4 * (N_PTS / 16), 256, 0, stream>>>(hn, out);
}
